// AdaptiveChebBlock_60722247631726
// MI455X (gfx1250) — matrix operand order repaired and checked by reading, not compiled, not run on hardware
//
#include <hip/hip_runtime.h>
#include <math.h>

// ---------------------------------------------------------------------------
// AdaptiveChebBlock for MI455X (gfx1250, wave32, WMMA).
// B=4, N=4096, D=H=128, K=3, TOPK=32, TELEPORT=0.1, LN_EPS=1e-5.
// Heavy parts (GEMM1, cosine-similarity, GEMM2) use v_wmma_f32_16x16x32_f16
// with fp32 accumulation. Sparse Chebyshev matvecs are gather/VALU bound.
// Workspace usage: ~40.2 MB.
// ---------------------------------------------------------------------------

typedef _Float16 h16;
typedef __attribute__((ext_vector_type(4)))  _Float16 v4h;
typedef __attribute__((ext_vector_type(8)))  _Float16 v8h;
typedef __attribute__((ext_vector_type(16))) _Float16 v16h;
typedef __attribute__((ext_vector_type(8)))  float    v8f;

#define NROWS   16384      // B*N
#define NBATCH  4096       // N
#define DFEAT   128        // D == H
#define HCATW   384        // H*K
#define TOPK_   32

static __device__ __forceinline__ v16h mkv16(v8h lo, v8h hi) {
  union { v16h v; v8h h[2]; } u;
  u.h[0] = lo; u.h[1] = hi;
  return u.v;
}

// A-operand (16x32 f16 chunk): lane l -> row (l&15); halves kbase+(l>>4)*8+{0..7}
// and kbase+16+(l>>4)*8+{0..7}.
static __device__ __forceinline__ v16h load_wmma_a(const h16* row, int kbase, int hi) {
  const v8h* p0 = reinterpret_cast<const v8h*>(row + kbase + hi * 8);
  const v8h* p1 = reinterpret_cast<const v8h*>(row + kbase + 16 + hi * 8);
  return mkv16(*p0, *p1);
}

// B-operand (32x16 f16 chunk) built from row-major source S with B[k][n]=S[n][k]:
// lane l -> source row (colbase + (l&15)); halves kbase+(l>>4)*8+{0..7}
// and kbase+16+(l>>4)*8+{0..7}.
static __device__ __forceinline__ v16h load_wmma_b(const h16* row, int kbase, int hi) {
  const v8h* p0 = reinterpret_cast<const v8h*>(row + kbase + hi * 8);
  const v8h* p1 = reinterpret_cast<const v8h*>(row + kbase + 16 + hi * 8);
  return mkv16(*p0, *p1);
}

static __device__ __forceinline__ float wave_sum(float v) {
#pragma unroll
  for (int o = 16; o > 0; o >>= 1) v += __shfl_xor(v, o, 32);
  return v;
}

// --------------------------- K0: weight prep -------------------------------
// w1t[n][k] = (f16) w1[k][n]   (128 x 128)
// w2t[n][k] = (f16) w2[k][n]   (128 x 384)
__global__ void k0_prep_weights(const float* __restrict__ w1,
                                const float* __restrict__ w2,
                                h16* __restrict__ w1t, h16* __restrict__ w2t) {
  int i = blockIdx.x * blockDim.x + threadIdx.x;
  if (i < 128 * 384) {
    int n = i / 384, k = i % 384;
    w2t[n * 384 + k] = (h16)w2[k * 128 + n];
  }
  if (i < 128 * 128) {
    int n = i >> 7, k = i & 127;
    w1t[n * 128 + k] = (h16)w1[k * 128 + n];
  }
}

// --------------------------- K1: LayerNorm ---------------------------------
// One wave per row (8 rows per 256-thread block).
__global__ void k1_layernorm(const float* __restrict__ x,
                             const float* __restrict__ g,
                             const float* __restrict__ b,
                             h16* __restrict__ xln) {
  int row  = blockIdx.x * 8 + (threadIdx.x >> 5);
  int lane = threadIdx.x & 31;
  const float4 xv = *reinterpret_cast<const float4*>(x + (size_t)row * DFEAT + lane * 4);
  float mu = wave_sum(xv.x + xv.y + xv.z + xv.w) * (1.f / 128.f);
  float d0 = xv.x - mu, d1 = xv.y - mu, d2 = xv.z - mu, d3 = xv.w - mu;
  float var = wave_sum(d0 * d0 + d1 * d1 + d2 * d2 + d3 * d3) * (1.f / 128.f);
  float rstd = rsqrtf(var + 1e-5f);
  const float4 gv = *reinterpret_cast<const float4*>(g + lane * 4);
  const float4 bv = *reinterpret_cast<const float4*>(b + lane * 4);
  v4h o = { (h16)(d0 * rstd * gv.x + bv.x), (h16)(d1 * rstd * gv.y + bv.y),
            (h16)(d2 * rstd * gv.z + bv.z), (h16)(d3 * rstd * gv.w + bv.w) };
  *reinterpret_cast<v4h*>(xln + (size_t)row * DFEAT + lane * 4) = o;
}

// --------------------------- K2: GEMM1 + GELU ------------------------------
// One wave per 16x16 output tile; K=128 -> 4 WMMA chunks. Exact-erf GELU.
__global__ void k2_gemm1_gelu(const h16* __restrict__ xln,
                              const h16* __restrict__ w1t,
                              const float* __restrict__ b1,
                              float* __restrict__ h32) {
  int tile = blockIdx.x * 4 + (threadIdx.x >> 5);
  int lane = threadIdx.x & 31;
  int cb = tile & 7, rb = tile >> 3;
  int n = lane & 15, hi = lane >> 4;
  const h16* arow = xln + (size_t)(rb * 16 + n) * DFEAT;
  const h16* brow = w1t + (size_t)(cb * 16 + n) * DFEAT;
  v8f c = {};
#pragma unroll
  for (int ch = 0; ch < 4; ++ch) {
    v16h a = load_wmma_a(arow, ch * 32, hi);
    v16h b = load_wmma_b(brow, ch * 32, hi);
    c = __builtin_amdgcn_wmma_f32_16x16x32_f16(false, a, false, b, (short)0, c, false, false);
  }
  int col = cb * 16 + n;
  float bias = b1[col];
#pragma unroll
  for (int v = 0; v < 8; ++v) {
    int row = rb * 16 + hi * 8 + v;
    float t = c[v] + bias;
    float gl = 0.5f * t * (1.f + erff(t * 0.70710678118654752f));
    h32[(size_t)row * DFEAT + col] = gl;
  }
}

// --------------------------- K3: row normalize -----------------------------
// xn = h/||h||2 (f16); also write T0 = f16(h) into hcat[:, 0:128].
__global__ void k3_rownorm(const float* __restrict__ h32,
                           h16* __restrict__ xn, h16* __restrict__ hcat) {
  int row  = blockIdx.x * 8 + (threadIdx.x >> 5);
  int lane = threadIdx.x & 31;
  const float4 hv = *reinterpret_cast<const float4*>(h32 + (size_t)row * DFEAT + lane * 4);
  float ss  = wave_sum(hv.x * hv.x + hv.y * hv.y + hv.z * hv.z + hv.w * hv.w);
  float inv = 1.f / fmaxf(sqrtf(ss), 1e-12f);
  v4h nn = { (h16)(hv.x * inv), (h16)(hv.y * inv), (h16)(hv.z * inv), (h16)(hv.w * inv) };
  *reinterpret_cast<v4h*>(xn + (size_t)row * DFEAT + lane * 4) = nn;
  v4h t0 = { (h16)hv.x, (h16)hv.y, (h16)hv.z, (h16)hv.w };
  *reinterpret_cast<v4h*>(hcat + (size_t)row * HCATW + lane * 4) = t0;
}

// --------------------------- K4: similarity + top-k ------------------------
// One 16-row block per workgroup (4 waves). Each wave streams 64 of the 256
// column tiles (4 WMMAs each), spills the 16x16 fp32 tile to its private LDS
// buffer, and lanes 0-15 maintain a running per-row top-32 (value,index) list.
// Tile buffer + lists are wave-private, so only wave-level ordering is needed
// inside the loop (LDS ops of one wave are in-order); a single __syncthreads
// precedes the cross-wave merge, which is done with full-wave argmax
// reductions (4 rows per wave).
__global__ void k4_simtopk(const h16* __restrict__ xn,
                           const float* __restrict__ log_tau,
                           unsigned* __restrict__ gidx,
                           float* __restrict__ gvals,
                           float* __restrict__ dm12) {
  __shared__ float    ltile[4][256];
  __shared__ float    lval[4][16][TOPK_];
  __shared__ unsigned lidx[4][16][TOPK_];

  int tid  = threadIdx.x;
  int w    = tid >> 5;
  int lane = tid & 31;
  int rb   = blockIdx.x;          // 16-row block id, 0..1023
  int batch = rb >> 8;            // 256 blocks per batch

  for (int t = tid; t < 4 * 16 * TOPK_; t += 128) {
    ((float*)lval)[t]    = 0.f;
    ((unsigned*)lidx)[t] = 0xFFFFFFFFu;
  }
  __syncthreads();

  int m = lane & 15, hi = lane >> 4;
  const h16* arow = xn + (size_t)(rb * 16 + m) * DFEAT;
  v16h a0 = load_wmma_a(arow, 0,  hi);
  v16h a1 = load_wmma_a(arow, 32, hi);
  v16h a2 = load_wmma_a(arow, 64, hi);
  v16h a3 = load_wmma_a(arow, 96, hi);

  float curmin = 0.f;
  int   minpos = 0;

  for (int ct = w; ct < 256; ct += 4) {
    int colbase = batch * NBATCH + ct * 16;
    const h16* brow = xn + (size_t)(colbase + m) * DFEAT;
    if (ct + 4 < 256)
      __builtin_prefetch((const void*)(brow + 64 * DFEAT), 0, 1);

    v8f c = {};
    c = __builtin_amdgcn_wmma_f32_16x16x32_f16(false, a0, false, load_wmma_b(brow, 0,  hi), (short)0, c, false, false);
    c = __builtin_amdgcn_wmma_f32_16x16x32_f16(false, a1, false, load_wmma_b(brow, 32, hi), (short)0, c, false, false);
    c = __builtin_amdgcn_wmma_f32_16x16x32_f16(false, a2, false, load_wmma_b(brow, 64, hi), (short)0, c, false, false);
    c = __builtin_amdgcn_wmma_f32_16x16x32_f16(false, a3, false, load_wmma_b(brow, 96, hi), (short)0, c, false, false);

    {
      int mb = hi * 8;
#pragma unroll
      for (int v = 0; v < 8; ++v) ltile[w][(mb + v) * 16 + m] = c[v];
    }
    // Tile buffer and lists are private to this wave; same-wave LDS ops are
    // in-order, so a compiler-level wave barrier is all that's required.
    __builtin_amdgcn_wave_barrier();

    if (lane < 16) {
      int r = lane;
      // Branch-free fast path: max of the 16 candidates first.
      float mx = 0.f;
#pragma unroll
      for (int cc = 0; cc < 16; ++cc) mx = fmaxf(mx, ltile[w][r * 16 + cc]);
      if (mx > curmin) {           // rare (~k*ln(N/k) insertions total per row)
        for (int cc = 0; cc < 16; ++cc) {
          float v = ltile[w][r * 16 + cc];  // relu implicit: curmin >= 0
          if (v > curmin) {
            lval[w][r][minpos] = v;
            lidx[w][r][minpos] = (unsigned)(colbase + cc);
            float mn = lval[w][r][0];
            int   mp = 0;
#pragma unroll
            for (int e = 1; e < TOPK_; ++e) {
              float q = lval[w][r][e];
              if (q < mn) { mn = q; mp = e; }
            }
            curmin = mn;
            minpos = mp;
          }
        }
      }
    }
    __builtin_amdgcn_wave_barrier();
  }

  __syncthreads();

  // Cross-wave merge: 128 candidates -> global top-32 per row. Each wave
  // handles 4 rows; per round, all 32 lanes scan 4 candidates each and an
  // argmax shfl_xor reduction (deterministic tie-break on candidate id)
  // picks the winner.
  {
    float tau = fmaxf(expf(log_tau[0]), 1e-3f);
    float self_add = 0.9f / tau + 0.1f;   // (1-TELEPORT)/tau + TELEPORT on diag
    for (int rr = 0; rr < 4; ++rr) {
      int r    = w * 4 + rr;
      int rowg = rb * 16 + r;
      float deg = 0.f;
      for (int s = 0; s < TOPK_; ++s) {
        float bv = 0.f;
        int   bq = 128;
#pragma unroll
        for (int t = 0; t < 4; ++t) {
          int q = lane * 4 + t;
          float v = lval[q >> 5][r][q & 31];
          if (v > bv) { bv = v; bq = q; }
        }
#pragma unroll
        for (int o = 16; o > 0; o >>= 1) {
          float ov = __shfl_xor(bv, o, 32);
          int   oq = __shfl_xor(bq, o, 32);
          if (ov > bv || (ov == bv && oq < bq)) { bv = ov; bq = oq; }
        }
        unsigned bidx = 0xFFFFFFFFu;
        float    fv   = 0.f;
        if (bv > 0.f) {
          bidx = lidx[bq >> 5][r][bq & 31];
          if (lane == (bq >> 2)) lval[bq >> 5][r][bq & 31] = -1.f;  // remove winner
          fv = 0.9f * bv / tau;
          if (bidx == (unsigned)rowg) fv += self_add;   // +I then teleport on diagonal
        }
        deg += fv;
        if (lane == 0) {
          gvals[(size_t)rowg * TOPK_ + s] = fv;
          gidx [(size_t)rowg * TOPK_ + s] = bidx;
        }
      }
      if (lane == 0) dm12[rowg] = rsqrtf(fmaxf(deg, 1e-6f));
    }
  }
}

// --------------------------- K5: sparse Chebyshev step ---------------------
// out = alpha * (Ahat @ in) + beta * base, wave per row; writes f16 slice of
// hcat and optionally an fp32 copy for the next step.
__global__ void k5_cheb_prop(const float* __restrict__ in32,
                             const float* __restrict__ base32,
                             const unsigned* __restrict__ gidx,
                             const float* __restrict__ gvals,
                             const float* __restrict__ dm12,
                             float alpha, float beta,
                             h16* __restrict__ out16, int coloff,
                             float* __restrict__ out32) {
  int row  = blockIdx.x * 8 + (threadIdx.x >> 5);
  int lane = threadIdx.x & 31;
  unsigned idx = gidx[(size_t)row * TOPK_ + lane];
  float    val = gvals[(size_t)row * TOPK_ + lane];
  float di = dm12[row];
  float w  = 0.f;
  int   jr = 0;
  if (idx != 0xFFFFFFFFu) { jr = (int)idx; w = val * di * dm12[idx]; }

  int f = lane * 4;
  float ax = 0.f, ay = 0.f, az = 0.f, aw = 0.f;
#pragma unroll 4
  for (int j = 0; j < TOPK_; ++j) {
    float wj = __shfl(w,  j, 32);
    int   rj = __shfl(jr, j, 32);
    const float4 hv = *reinterpret_cast<const float4*>(in32 + (size_t)rj * DFEAT + f);
    ax += wj * hv.x; ay += wj * hv.y; az += wj * hv.z; aw += wj * hv.w;
  }
  float ox = alpha * ax, oy = alpha * ay, oz = alpha * az, ow = alpha * aw;
  if (base32) {
    const float4 bs = *reinterpret_cast<const float4*>(base32 + (size_t)row * DFEAT + f);
    ox += beta * bs.x; oy += beta * bs.y; oz += beta * bs.z; ow += beta * bs.w;
  }
  v4h o16 = { (h16)ox, (h16)oy, (h16)oz, (h16)ow };
  *reinterpret_cast<v4h*>(out16 + (size_t)row * HCATW + coloff + f) = o16;
  if (out32) {
    float4 o; o.x = ox; o.y = oy; o.z = oz; o.w = ow;
    *reinterpret_cast<float4*>(out32 + (size_t)row * DFEAT + f) = o;
  }
}

// --------------------------- K6: GEMM2 + residual --------------------------
// One wave per 16x16 output tile; K=384 -> 12 WMMA chunks.
__global__ void k6_gemm2_out(const h16* __restrict__ hcat,
                             const h16* __restrict__ w2t,
                             const float* __restrict__ b2,
                             const float* __restrict__ x,
                             const float* __restrict__ gate,
                             float* __restrict__ out) {
  int tile = blockIdx.x * 4 + (threadIdx.x >> 5);
  int lane = threadIdx.x & 31;
  int cb = tile & 7, rb = tile >> 3;
  int n = lane & 15, hi = lane >> 4;
  const h16* arow = hcat + (size_t)(rb * 16 + n) * HCATW;
  const h16* brow = w2t  + (size_t)(cb * 16 + n) * HCATW;
  v8f c = {};
#pragma unroll
  for (int ch = 0; ch < 12; ++ch) {
    v16h a = load_wmma_a(arow, ch * 32, hi);
    v16h b = load_wmma_b(brow, ch * 32, hi);
    c = __builtin_amdgcn_wmma_f32_16x16x32_f16(false, a, false, b, (short)0, c, false, false);
  }
  int col = cb * 16 + n;
  float bias = b2[col];
  float tg = tanhf(gate[0]);
#pragma unroll
  for (int v = 0; v < 8; ++v) {
    int row = rb * 16 + hi * 8 + v;
    float y = c[v] + bias;
    out[(size_t)row * DFEAT + col] = x[(size_t)row * DFEAT + col] + tg * y;
  }
}

// --------------------------- host launcher ---------------------------------
extern "C" void kernel_launch(void* const* d_in, const int* in_sizes, int n_in,
                              void* d_out, int out_size, void* d_ws, size_t ws_size,
                              hipStream_t stream) {
  (void)in_sizes; (void)n_in; (void)out_size; (void)ws_size;

  const float* x       = (const float*)d_in[0];
  const float* ln_g    = (const float*)d_in[1];
  const float* ln_b    = (const float*)d_in[2];
  const float* w1      = (const float*)d_in[3];
  const float* b1      = (const float*)d_in[4];
  const float* w2      = (const float*)d_in[5];
  const float* b2      = (const float*)d_in[6];
  const float* log_tau = (const float*)d_in[7];
  const float* gate    = (const float*)d_in[8];
  float* out = (float*)d_out;

  char* ws = (char*)d_ws;
  // Workspace layout (bytes), total ~40.2 MB.
  h16*      w1t   = (h16*)(ws + 0);                       //   32768
  h16*      w2t   = (h16*)(ws + 32768);                   //   98304
  h16*      xln   = (h16*)(ws + 131072);                  // 4194304
  h16*      xn    = (h16*)(ws + 4325376);                 // 4194304
  float*    h32   = (float*)(ws + 8519680);               // 8388608
  float*    t1    = (float*)(ws + 16908288);              // 8388608
  h16*      hcat  = (h16*)(ws + 25296896);                // 12582912
  unsigned* gidx  = (unsigned*)(ws + 37879808);           // 2097152
  float*    gvals = (float*)(ws + 39976960);              // 2097152
  float*    dm12  = (float*)(ws + 42074112);              //   65536

  // K0: weight transpose/convert (128*384 = 49152 elements)
  k0_prep_weights<<<192, 256, 0, stream>>>(w1, w2, w1t, w2t);

  // K1: LayerNorm (wave per row)
  k1_layernorm<<<NROWS / 8, 256, 0, stream>>>(x, ln_g, ln_b, xln);

  // K2: GEMM1 + GELU (8192 tiles, 4 waves/WG)
  k2_gemm1_gelu<<<2048, 128, 0, stream>>>(xln, w1t, b1, h32);

  // K3: row L2-normalize; write T0 f16
  k3_rownorm<<<NROWS / 8, 256, 0, stream>>>(h32, xn, hcat);

  // K4: cosine similarity + top-32 (one 16-row block per WG)
  k4_simtopk<<<NROWS / 16, 128, 0, stream>>>(xn, log_tau, gidx, gvals, dm12);

  // K5: T1 = Ahat @ T0
  k5_cheb_prop<<<NROWS / 8, 256, 0, stream>>>(h32, nullptr, gidx, gvals, dm12,
                                              1.0f, 0.0f, hcat, 128, t1);
  // K5: T2 = 2 * Ahat @ T1 - T0
  k5_cheb_prop<<<NROWS / 8, 256, 0, stream>>>(t1, h32, gidx, gvals, dm12,
                                              2.0f, -1.0f, hcat, 256, nullptr);

  // K6: y = hcat @ w2 + b2; out = x + tanh(gate) * y
  k6_gemm2_out<<<2048, 128, 0, stream>>>(hcat, w2t, b2, x, gate, out);
}
